// FP8Linear_35364760715641
// MI455X (gfx1250) — 8-bit matrix operand order repaired and checked by reading, compiled, not run on hardware
//
#include <hip/hip_runtime.h>
#include <hip/hip_bf16.h>

typedef __attribute__((ext_vector_type(16))) int   v16i;
typedef __attribute__((ext_vector_type(8)))  float v8f;
typedef __attribute__((ext_vector_type(4)))  int   v4i;

typedef __attribute__((address_space(1))) v4i* gv4p;
typedef __attribute__((address_space(3))) v4i* lv4p;

#define FP8_MAX 448.0f
#define BLK 128

#if __has_builtin(__builtin_amdgcn_global_load_async_to_lds_b128)
#define HAVE_ASYNC_LDS 1
#else
#define HAVE_ASYNC_LDS 0
#endif

__device__ __forceinline__ void wait_async0() {
#if HAVE_ASYNC_LDS
#if __has_builtin(__builtin_amdgcn_s_wait_asynccnt)
    __builtin_amdgcn_s_wait_asynccnt(0);
#else
    asm volatile("s_wait_asynccnt 0x0" ::: "memory");
#endif
#endif
}

__device__ __forceinline__ unsigned int enc_e4m3(float x) {
    unsigned int u    = __float_as_uint(x);
    unsigned int sign = (u >> 24) & 0x80u;
    unsigned int au   = u & 0x7FFFFFFFu;
    if (au > 0x43E00000u) au = 0x43E00000u;
    unsigned int lsb = (au >> 20) & 1u;
    au += 0x0007FFFFu + lsb;
    au &= 0xFFF00000u;
    int e = (int)(au >> 23) - 127;
    unsigned int q;
    if (e < -9) {
        q = 0u;
    } else if (e < -6) {
        float ax = __uint_as_float(au);
        q = (unsigned int)(ax * 512.0f + 0.5f);
    } else {
        q = (unsigned int)(((e + 7) << 3) | ((au >> 20) & 7u));
    }
    return q | sign;
}

__device__ __forceinline__ unsigned int pack4_e4m3(float x, float y, float z, float w) {
#if __has_builtin(__builtin_amdgcn_cvt_pk_fp8_f32)
    int lo = __builtin_amdgcn_cvt_pk_fp8_f32(x, y, 0, false);
    int p  = __builtin_amdgcn_cvt_pk_fp8_f32(z, w, lo, true);
    return (unsigned int)p;
#else
    return enc_e4m3(x) | (enc_e4m3(y) << 8) | (enc_e4m3(z) << 16) | (enc_e4m3(w) << 24);
#endif
}

__global__ __launch_bounds__(256) void quantA_kernel(
        const float* __restrict__ x, unsigned char* __restrict__ aq,
        float* __restrict__ ascale, int M, int K) {
    const int nb   = K >> 7;
    const int wave = (int)((blockIdx.x * blockDim.x + threadIdx.x) >> 5);
    const int lane = threadIdx.x & 31;
    const int row  = wave / nb;
    const int kb   = wave % nb;
    if (row >= M) return;

    const size_t base = (size_t)row * K + (size_t)kb * BLK + lane * 4;
    const float4 v = *(const float4*)(x + base);
    float m = fmaxf(fmaxf(fabsf(v.x), fabsf(v.y)), fmaxf(fabsf(v.z), fabsf(v.w)));
    #pragma unroll
    for (int off = 16; off > 0; off >>= 1)
        m = fmaxf(m, __shfl_xor(m, off, 32));
    m = fmaxf(m, 1e-4f);
    const float inv = FP8_MAX / m;

    *(unsigned int*)(aq + base) =
        pack4_e4m3(v.x * inv, v.y * inv, v.z * inv, v.w * inv);
    if (lane == 0) ascale[(size_t)row * nb + kb] = m / FP8_MAX;
}

__global__ __launch_bounds__(256) void quantW_kernel(
        const float* __restrict__ w, unsigned char* __restrict__ wq,
        float* __restrict__ wscale, int N, int K) {
    __shared__ float red[256];
    const int kb  = blockIdx.x;
    const int nbk = blockIdx.y;
    const int t   = threadIdx.x;
    const int row  = nbk * BLK + (t >> 1);
    const int col0 = kb * BLK + (t & 1) * 64;

    const float4* g = (const float4*)(w + (size_t)row * K + col0);
    float4 v[16];
    float m = 0.0f;
    #pragma unroll
    for (int i = 0; i < 16; ++i) {
        v[i] = g[i];
        m = fmaxf(m, fmaxf(fmaxf(fabsf(v[i].x), fabsf(v[i].y)),
                           fmaxf(fabsf(v[i].z), fabsf(v[i].w))));
    }
    red[t] = m;
    __syncthreads();
    for (int s = 128; s > 0; s >>= 1) {
        if (t < s) red[t] = fmaxf(red[t], red[t + s]);
        __syncthreads();
    }
    const float amax = fmaxf(red[0], 1e-4f);
    const float inv  = FP8_MAX / amax;
    if (t == 0) wscale[(size_t)nbk * (K >> 7) + kb] = amax / FP8_MAX;

    unsigned int* o = (unsigned int*)(wq + (size_t)row * K + col0);
    #pragma unroll
    for (int i = 0; i < 16; ++i)
        o[i] = pack4_e4m3(v[i].x * inv, v[i].y * inv, v[i].z * inv, v[i].w * inv);
}

#define BM 64
#define BN 128
#define LDT 144
#define NBK 16

__global__ __launch_bounds__(256) void gemm_fp8_kernel(
        const unsigned char* __restrict__ aq, const float* __restrict__ ascale,
        const unsigned char* __restrict__ wq, const float* __restrict__ wscale,
        __hip_bfloat16* __restrict__ out, int M, int N, int K) {
    __shared__ __align__(16) unsigned char As[2][BM * LDT];
    __shared__ __align__(16) unsigned char Bs[2][BN * LDT];
    __shared__ float Ssc[NBK][BM];

    const int tid  = threadIdx.x;
    const int wave = tid >> 5;
    const int lane = tid & 31;
    const int half = lane >> 4;
    const int lr   = lane & 15;
    const int wm   = wave >> 2;
    const int wn   = wave & 3;
    const int block_m = blockIdx.y * BM;
    const int block_n = blockIdx.x * BN;
    const int nb   = K >> 7;
    const int nblk = block_n >> 7;

    for (int i = tid; i < NBK * BM; i += 256) {
        const int kb = i >> 6;
        const int r  = i & (BM - 1);
        Ssc[kb][r] = ascale[(size_t)(block_m + r) * nb + kb];
    }

    const int rA = tid >> 2, cA = (tid & 3) * 32;
    const int rB = tid >> 1, cB = (tid & 1) * 64;
    const size_t aOff = (size_t)(block_m + rA) * K + cA;
    const size_t wOff = (size_t)(block_n + rB) * K + cB;

    auto stage = [&](int kb, int buf) {
        const unsigned char* ga = aq + aOff + (size_t)kb * BLK;
        const unsigned char* gw = wq + wOff + (size_t)kb * BLK;
        unsigned char* la = &As[buf][rA * LDT + cA];
        unsigned char* lb = &Bs[buf][rB * LDT + cB];
#if HAVE_ASYNC_LDS
        #pragma unroll
        for (int i = 0; i < 2; ++i)
            __builtin_amdgcn_global_load_async_to_lds_b128(
                (gv4p)(ga + i * 16), (lv4p)(la + i * 16), 0, 0);
        #pragma unroll
        for (int i = 0; i < 4; ++i)
            __builtin_amdgcn_global_load_async_to_lds_b128(
                (gv4p)(gw + i * 16), (lv4p)(lb + i * 16), 0, 0);
#else
        int4 ra0 = ((const int4*)ga)[0];
        int4 ra1 = ((const int4*)ga)[1];
        int4 rb0 = ((const int4*)gw)[0];
        int4 rb1 = ((const int4*)gw)[1];
        int4 rb2 = ((const int4*)gw)[2];
        int4 rb3 = ((const int4*)gw)[3];
        ((int4*)la)[0] = ra0; ((int4*)la)[1] = ra1;
        ((int4*)lb)[0] = rb0; ((int4*)lb)[1] = rb1;
        ((int4*)lb)[2] = rb2; ((int4*)lb)[3] = rb3;
#endif
    };

    v8f acc[2][2] = {};

    stage(0, 0);
    wait_async0();
    __syncthreads();

    for (int kb = 0; kb < nb; ++kb) {
        const int  buf = kb & 1;
        const bool pf  = (kb + 1) < nb;

#if HAVE_ASYNC_LDS
        if (pf) stage(kb + 1, buf ^ 1);
#endif
        const float ws_k = wscale[(size_t)nblk * nb + kb];

        v16i  afrag[2];
        float asc[2][8];
        #pragma unroll
        for (int mt = 0; mt < 2; ++mt) {
            const int am = wm * 32 + mt * 16;
            unsigned long long* a64 = (unsigned long long*)&afrag[mt];
            #pragma unroll
            for (int h2 = 0; h2 < 2; ++h2)
                #pragma unroll
                for (int c4 = 0; c4 < 4; ++c4)
                    a64[h2 * 4 + c4] = *(const unsigned long long*)
                        &As[buf][(am + lr) * LDT + h2 * 64 + (c4 >> 1) * 32 + half * 16 + (c4 & 1) * 8];
            const float4 s0 = *(const float4*)&Ssc[kb][am + half * 8];
            const float4 s1 = *(const float4*)&Ssc[kb][am + half * 8 + 4];
            asc[mt][0] = s0.x * ws_k; asc[mt][1] = s0.y * ws_k;
            asc[mt][2] = s0.z * ws_k; asc[mt][3] = s0.w * ws_k;
            asc[mt][4] = s1.x * ws_k; asc[mt][5] = s1.y * ws_k;
            asc[mt][6] = s1.z * ws_k; asc[mt][7] = s1.w * ws_k;
        }

        #pragma unroll
        for (int nt = 0; nt < 2; ++nt) {
            const int bn = wn * 32 + nt * 16;
            v16i bfrag;
            int4* b4 = (int4*)&bfrag;
            #pragma unroll
            for (int q = 0; q < 4; ++q)
                b4[q] = *(const int4*)&Bs[buf][(bn + lr) * LDT + q * 32 + half * 16];

            #pragma unroll
            for (int mt = 0; mt < 2; ++mt) {
                v8f zero = {};
                v8f d = __builtin_amdgcn_wmma_f32_16x16x128_fp8_fp8(
                            afrag[mt], bfrag, (short)0, zero, false, false);
                #pragma unroll
                for (int r = 0; r < 8; ++r)
                    acc[mt][nt][r] += d[r] * asc[mt][r];
            }
        }

#if !HAVE_ASYNC_LDS
        if (pf) stage(kb + 1, buf ^ 1);
#endif
        wait_async0();
        __syncthreads();
    }

    #pragma unroll
    for (int mt = 0; mt < 2; ++mt)
        #pragma unroll
        for (int nt = 0; nt < 2; ++nt) {
            const int m0 = block_m + wm * 32 + mt * 16 + half * 8;
            const int n0 = block_n + wn * 32 + nt * 16 + lr;
            #pragma unroll
            for (int r = 0; r < 8; ++r)
                out[(size_t)(m0 + r) * N + n0] = __float2bfloat16(acc[mt][nt][r]);
        }
}

extern "C" void kernel_launch(void* const* d_in, const int* in_sizes, int n_in,
                              void* d_out, int out_size, void* d_ws, size_t ws_size,
                              hipStream_t stream) {
    const float* x = (const float*)d_in[0];
    const float* w = (const float*)d_in[1];
    const int K = 2048;
    const int M = in_sizes[0] / K;
    const int N = in_sizes[1] / K;
    const int nb = K / BLK;

    unsigned char* aq = (unsigned char*)d_ws;
    size_t off = (size_t)M * K;
    unsigned char* wqp = (unsigned char*)d_ws + off;
    off += (size_t)N * K;
    off = (off + 255) & ~(size_t)255;
    float* as = (float*)((char*)d_ws + off);
    off += (size_t)M * nb * sizeof(float);
    off = (off + 255) & ~(size_t)255;
    float* wsp = (float*)((char*)d_ws + off);

    quantA_kernel<<<(M * nb) / 8, 256, 0, stream>>>(x, aq, as, M, K);
    quantW_kernel<<<dim3(K / BLK, N / BLK), 256, 0, stream>>>(w, wqp, wsp, N, K);
    gemm_fp8_kernel<<<dim3(N / BN, M / BM), 256, 0, stream>>>(
        aq, as, wqp, wsp, (__hip_bfloat16*)d_out, M, N, K);
}
